// IntraOptionPolicy_32693291057943
// MI455X (gfx1250) — compile-verified, matrix operand order checked by reading, not run on hardware
//
#include <hip/hip_runtime.h>
#include <hip/hip_bf16.h>

typedef __attribute__((ext_vector_type(16))) __bf16 v16bf;
typedef __attribute__((ext_vector_type(8)))  __bf16 v8bf;
typedef __attribute__((ext_vector_type(8)))  float  v8f;

// ---- problem sizes (compile-time) ----
constexpr int B_   = 8192;
constexpr int OBS  = 1024;
constexpr int HID  = 256;
constexpr int U_   = 8;
constexpr int OPT  = 8;
constexpr int ACT  = 18;
constexpr int INV  = 400;

// =====================================================================
// Generic bf16 WMMA GEMM: C[M,N] = A[M,K](f32) * B[K,N](f32), f32 accum.
// Macro tile 128x128, BK=32, 256 threads (8 waves), each wave does a
// 64x32 sub-tile = 4x2 WMMA tiles of 16x16. Double-buffered LDS.
// =====================================================================
struct GemmP {
  const float* A; const float* Bm; float* Out;
  const float* bias; const float* aux0; const float* aux1; const float* msk;
  const int* opt;
  int M, N, K, lda, ldb, ldo;
  long a_zs, b_zs, o_zs, bias_zs;   // per-blockIdx.z strides
};

enum { EPI_STORE = 0, EPI_STORE_BIAS = 1, EPI_STORE_BIAS_RELU = 2,
       EPI_QSEL = 3, EPI_SCALE = 4, EPI_ACC_BIAS = 5, EPI_HCOMM = 6 };

__device__ __forceinline__ v16bf load_frag(const __bf16* rowbase, int khalf) {
  union { v16bf v; v8bf h[2]; } f;
  f.h[0] = *(const v8bf*)(rowbase + khalf);        // K = khalf .. khalf+7
  f.h[1] = *(const v8bf*)(rowbase + 16 + khalf);   // K = 16+khalf .. +7
  return f.v;
}

template<int EPI>
__global__ __launch_bounds__(256) void gemm_bf16_wmma(GemmP p) {
  __shared__ v8bf AlsV[2][512];   // 128 rows x 32 k (bf16), double buffered
  __shared__ v8bf BlsV[2][512];   // 128 n    x 32 k (B transposed, bf16)

  const int tid   = threadIdx.x;
  const int lane  = tid & 31;
  const int w     = tid >> 5;
  const int waveM = w >> 2;            // 0..1
  const int waveN = w & 3;             // 0..3
  const int z     = blockIdx.z;
  const int tileM = blockIdx.y * 128;
  const int tileN = blockIdx.x * 128;
  const float* A  = p.A  + (long)z * p.a_zs;
  const float* Bg = p.Bm + (long)z * p.b_zs;

  v8f acc[4][2];
  const v8f vzero = {0.f,0.f,0.f,0.f,0.f,0.f,0.f,0.f};
#pragma unroll
  for (int mt = 0; mt < 4; ++mt)
#pragma unroll
    for (int nt = 0; nt < 2; ++nt) acc[mt][nt] = vzero;

  const int khalf = (lane & 16) ? 8 : 0;

  // staging assignment: A -> thread = half row (16 consecutive k)
  //                     B -> thread = one column (16 consecutive k, transposed)
  const int ar = tid >> 1;             // 0..127 (A row)
  const int ak = (tid & 1) * 16;       // 0 / 16 (A k-half)
  const int bn = tid & 127;            // 0..127 (B column)
  const int bk = (tid >> 7) * 16;      // 0 / 16 (B k-half)
  float ra[16], rb[16];

  auto fetch = [&](int k0) {
    const int gm = tileM + ar;
    const bool kfull = (k0 + 32 <= p.K);
    if (kfull && gm < p.M) {           // vector fast path (16B loads)
      const float4* av = (const float4*)(A + (long)gm * p.lda + k0 + ak);
#pragma unroll
      for (int j = 0; j < 4; ++j) {
        float4 f = av[j];
        ra[4*j+0] = f.x; ra[4*j+1] = f.y; ra[4*j+2] = f.z; ra[4*j+3] = f.w;
      }
    } else {
#pragma unroll
      for (int j = 0; j < 16; ++j) {
        int gk = k0 + ak + j;
        ra[j] = (gm < p.M && gk < p.K) ? A[(long)gm * p.lda + gk] : 0.f;
      }
    }
    const int gn = tileN + bn;
    if (kfull && gn < p.N) {           // coalesced-across-threads column gather
      const float* bp = Bg + (long)(k0 + bk) * p.ldb + gn;
#pragma unroll
      for (int j = 0; j < 16; ++j) rb[j] = bp[(long)j * p.ldb];
    } else {
#pragma unroll
      for (int j = 0; j < 16; ++j) {
        int gk = k0 + bk + j;
        rb[j] = (gk < p.K && gn < p.N) ? Bg[(long)gk * p.ldb + gn] : 0.f;
      }
    }
  };

  auto stage = [&](int buf) {
    __bf16* Als = (__bf16*)AlsV[buf];
    __bf16* Bls = (__bf16*)BlsV[buf];
    v8bf va0, va1, vb0, vb1;
#pragma unroll
    for (int j = 0; j < 8; ++j) {
      va0[j] = (__bf16)ra[j];  va1[j] = (__bf16)ra[8 + j];
      vb0[j] = (__bf16)rb[j];  vb1[j] = (__bf16)rb[8 + j];
    }
    *(v8bf*)(Als + ar * 32 + ak)     = va0;
    *(v8bf*)(Als + ar * 32 + ak + 8) = va1;
    *(v8bf*)(Bls + bn * 32 + bk)     = vb0;
    *(v8bf*)(Bls + bn * 32 + bk + 8) = vb1;
  };

  const int nsteps = (p.K + 31) >> 5;
  fetch(0);
  stage(0);
  __syncthreads();
  int cur = 0;
  for (int s = 0; s < nsteps; ++s) {
    const bool more = (s + 1 < nsteps);
    if (more) fetch((s + 1) * 32);     // global loads in flight during WMMA

    const __bf16* Als = (const __bf16*)AlsV[cur];
    const __bf16* Bls = (const __bf16*)BlsV[cur];
    v16bf afr[4], bfr[2];
#pragma unroll
    for (int mt = 0; mt < 4; ++mt)
      afr[mt] = load_frag(&Als[(waveM * 64 + mt * 16 + (lane & 15)) * 32], khalf);
#pragma unroll
    for (int nt = 0; nt < 2; ++nt)
      bfr[nt] = load_frag(&Bls[(waveN * 32 + nt * 16 + (lane & 15)) * 32], khalf);
#pragma unroll
    for (int mt = 0; mt < 4; ++mt)
#pragma unroll
      for (int nt = 0; nt < 2; ++nt)
        acc[mt][nt] = __builtin_amdgcn_wmma_f32_16x16x32_bf16(
            false, afr[mt], false, bfr[nt], (short)0, acc[mt][nt], false, false);

    if (more) stage(cur ^ 1);
    __syncthreads();
    cur ^= 1;
  }

  // ---- epilogue ----
  float* Out_eff = p.Out + (long)z * p.o_zs;
#pragma unroll
  for (int mt = 0; mt < 4; ++mt) {
#pragma unroll
    for (int nt = 0; nt < 2; ++nt) {
      int n = tileN + waveN * 32 + nt * 16 + (lane & 15);
#pragma unroll
      for (int r = 0; r < 8; ++r) {
        int m = tileM + waveM * 64 + mt * 16 + r + ((lane & 16) >> 1);
        if (m < p.M && n < p.N) {
          float c = acc[mt][nt][r];
          if constexpr (EPI == EPI_STORE) {
            Out_eff[(long)m * p.ldo + n] = c;
          } else if constexpr (EPI == EPI_STORE_BIAS) {
            const float* be = p.bias + (long)z * p.bias_zs;
            Out_eff[(long)m * p.ldo + n] = c + be[n];
          } else if constexpr (EPI == EPI_STORE_BIAS_RELU) {
            const float* be = p.bias + (long)z * p.bias_zs;
            float v = c + be[n];
            Out_eff[(long)m * p.ldo + n] = v > 0.f ? v : 0.f;
          } else if constexpr (EPI == EPI_QSEL) {
            // keep only columns of the per-row selected option; add b_q
            int o = n >> 6, q = n & 63;
            if (o == p.opt[m]) {
              const float* be = p.bias + (long)z * p.bias_zs;  // b_q + u*64
              Out_eff[(long)m * p.ldo + q] = c + be[o * 512 + q];
            }
          } else if constexpr (EPI == EPI_SCALE) {
            // pre1 = coef[b,u] * (v0 @ W_i2h[u])
            Out_eff[(long)m * p.ldo + n] = c * p.aux0[(long)m * 8 + z];
          } else if constexpr (EPI == EPI_ACC_BIAS) {
            const float* be = p.bias + (long)z * p.bias_zs;   // b_lstm[u]
            float* po = &Out_eff[(long)m * p.ldo + n];
            *po = *po + c + be[n];
          } else if constexpr (EPI == EPI_HCOMM) {
            // h_comm = ctx@Wo + h_b ; hy = m*h_comm + (1-m)*hx
            long bi = (long)m * 2048 + (long)z * 256 + n;
            float hcomm = c + p.aux0[bi];
            float mv = p.msk[(long)m * 8 + z];
            Out_eff[(long)m * p.ldo + n] = mv * hcomm + (1.f - mv) * p.aux1[bi];
          }
        }
      }
    }
  }
}

// =====================================================================
// Small VALU kernels
// =====================================================================
__device__ __forceinline__ float sigf(float x) { return 1.f / (1.f + __expf(-x)); }

// reshape W_q[o,u,h,q] -> Wqr[u][h][o*64+q]
__global__ void prep_wq_kernel(const float* __restrict__ Wq, float* __restrict__ Wqr) {
  int i = blockIdx.x * blockDim.x + threadIdx.x;
  if (i >= 8 * 256 * 512) return;
  int u = i >> 17, rem = i & 131071, h = rem >> 9, n = rem & 511;
  int o = n >> 6, q = n & 63;
  Wqr[i] = Wq[(((long)o * 8 + u) * 256 + h) * 64 + q];
}

// reshape pi_w[o,h,a] -> piR[h][o*18+a]
__global__ void prep_pi_kernel(const float* __restrict__ piw, float* __restrict__ pir) {
  int i = blockIdx.x * blockDim.x + threadIdx.x;
  if (i >= 256 * 144) return;
  int h = i / 144, n = i % 144, o = n / 18, a = n % 18;
  pir[i] = piw[((long)o * 256 + h) * 18 + a];
}

// scores[b,u] = dot64(qsel[b,u,:], k0[b,:]) / 8
__global__ void score_kernel(const float* __restrict__ qsel, const float* __restrict__ k0,
                             float* __restrict__ sc) {
  int i = blockIdx.x * blockDim.x + threadIdx.x;
  if (i >= B_ * U_) return;
  int b = i >> 3;
  const float* q = qsel + (long)i * 64;
  const float* k = k0 + (long)b * 64;
  float s = 0.f;
#pragma unroll 8
  for (int t = 0; t < 64; ++t) s += q[t] * k[t];
  sc[i] = s * 0.125f;
}

// top-k(4 of 8) mask (stable ties like lax.top_k) + coef = sigmoid(s)*mask
__global__ void topk_kernel(const float* __restrict__ sc, float* __restrict__ coef,
                            float* __restrict__ maskf) {
  int b = blockIdx.x * blockDim.x + threadIdx.x;
  if (b >= B_) return;
  float s[8];
#pragma unroll
  for (int u = 0; u < 8; ++u) s[u] = sc[b * 8 + u];
#pragma unroll
  for (int u = 0; u < 8; ++u) {
    int rank = 0;
#pragma unroll
    for (int u2 = 0; u2 < 8; ++u2)
      rank += (s[u2] > s[u]) || (s[u2] == s[u] && u2 < u);
    float m = (rank < 4) ? 1.f : 0.f;
    maskf[b * 8 + u] = m;
    coef[b * 8 + u]  = sigf(s[u]) * m;
  }
}

// LSTM gates: pre[b,u,1024] -> h_b[b,u,256], cy written to output
__global__ void lstm_kernel(const float* __restrict__ pre, const float* __restrict__ cx,
                            const float* __restrict__ maskf, float* __restrict__ hb,
                            float* __restrict__ cy) {
  long i = (long)blockIdx.x * blockDim.x + threadIdx.x;
  if (i >= (long)B_ * U_ * HID) return;
  int h = (int)(i & 255);
  long base = (i >> 8) * 1024;
  float iv = pre[base + h], fv = pre[base + 256 + h];
  float ov = pre[base + 512 + h], gv = pre[base + 768 + h];
  float cxv = cx[i];
  float cn = sigf(fv) * cxv + sigf(iv) * tanhf(gv);
  float hn = sigf(ov) * tanhf(cn);
  hb[i] = hn;
  float mv = maskf[i >> 8];
  cy[i] = mv * cn + (1.f - mv) * cxv;
}

// per-batch comm attention: 4 heads x (8x8) with K=32, masked after softmax
__global__ void attn_kernel(const float* __restrict__ qc, const float* __restrict__ kc,
                            const float* __restrict__ vc, const float* __restrict__ maskf,
                            float* __restrict__ ctx) {
  int b = blockIdx.x * 8 + (threadIdx.x >> 5);
  if (b >= B_) return;
  int lane = threadIdx.x & 31;
  int h = lane >> 3, iu = lane & 7;           // one (head,row) per lane
  float qrow[32];
  const float* qp = qc + ((long)(b * 8 + iu) * 128 + h * 32);
#pragma unroll
  for (int q = 0; q < 32; ++q) qrow[q] = qp[q];
  float sc[8];
#pragma unroll
  for (int j = 0; j < 8; ++j) {
    const float* kp = kc + ((long)(b * 8 + j) * 128 + h * 32);
    float d = 0.f;
#pragma unroll
    for (int q = 0; q < 32; ++q) d += qrow[q] * kp[q];
    sc[j] = d * 0.17677669529663687f;         // 1/sqrt(32)
  }
  float mx = sc[0];
#pragma unroll
  for (int j = 1; j < 8; ++j) mx = fmaxf(mx, sc[j]);
  float sum = 0.f;
#pragma unroll
  for (int j = 0; j < 8; ++j) { sc[j] = __expf(sc[j] - mx); sum += sc[j]; }
  float inv = 1.f / sum;
#pragma unroll
  for (int j = 0; j < 8; ++j) sc[j] = sc[j] * inv * maskf[b * 8 + j];
  float* cp = ctx + ((long)(b * 8 + iu) * 512 + h * 128);
  for (int d = 0; d < 128; ++d) {
    float o = 0.f;
#pragma unroll
    for (int j = 0; j < 8; ++j) o += sc[j] * vc[(long)(b * 8 + j) * 512 + h * 128 + d];
    cp[d] = o;
  }
}

// hsum[b,h] = sum_u hy[b,u,h]
__global__ void hsum_kernel(const float* __restrict__ hy, float* __restrict__ hs) {
  long i = (long)blockIdx.x * blockDim.x + threadIdx.x;
  if (i >= (long)B_ * HID) return;
  int b = (int)(i >> 8), h = (int)(i & 255);
  float s = 0.f;
#pragma unroll
  for (int u = 0; u < 8; ++u) s += hy[((long)b * 8 + u) * 256 + h];
  hs[i] = s;
}

// softmax over ACT=18 per (b,o)
__global__ void softmax18_kernel(const float* __restrict__ logits, float* __restrict__ out) {
  int i = blockIdx.x * blockDim.x + threadIdx.x;
  if (i >= B_ * OPT) return;
  long base = (long)(i >> 3) * 144 + (i & 7) * 18;
  float mx = logits[base];
  for (int a = 1; a < 18; ++a) mx = fmaxf(mx, logits[base + a]);
  float sum = 0.f, e[18];
  for (int a = 0; a < 18; ++a) { e[a] = __expf(logits[base + a] - mx); sum += e[a]; }
  float inv = 1.f / sum;
  for (int a = 0; a < 18; ++a) out[base + a] = e[a] * inv;
}

// =====================================================================
// host side
// =====================================================================
static void gemm_go(int epi, const GemmP& p, int gx, int gy, int gz, hipStream_t s) {
  dim3 g(gx, gy, gz), b(256);
  switch (epi) {
    case EPI_STORE:           gemm_bf16_wmma<EPI_STORE><<<g, b, 0, s>>>(p); break;
    case EPI_STORE_BIAS:      gemm_bf16_wmma<EPI_STORE_BIAS><<<g, b, 0, s>>>(p); break;
    case EPI_STORE_BIAS_RELU: gemm_bf16_wmma<EPI_STORE_BIAS_RELU><<<g, b, 0, s>>>(p); break;
    case EPI_QSEL:            gemm_bf16_wmma<EPI_QSEL><<<g, b, 0, s>>>(p); break;
    case EPI_SCALE:           gemm_bf16_wmma<EPI_SCALE><<<g, b, 0, s>>>(p); break;
    case EPI_ACC_BIAS:        gemm_bf16_wmma<EPI_ACC_BIAS><<<g, b, 0, s>>>(p); break;
    case EPI_HCOMM:           gemm_bf16_wmma<EPI_HCOMM><<<g, b, 0, s>>>(p); break;
  }
}

extern "C" void kernel_launch(void* const* d_in, const int* in_sizes, int n_in,
                              void* d_out, int out_size, void* d_ws, size_t ws_size,
                              hipStream_t stream) {
  (void)in_sizes; (void)n_in; (void)out_size; (void)ws_size;
  const float* x      = (const float*)d_in[0];
  const float* hx     = (const float*)d_in[1];
  const float* cx     = (const float*)d_in[2];
  const int*   option = (const int*)d_in[3];
  const float* W_pre  = (const float*)d_in[4];
  const float* b_pre  = (const float*)d_in[5];
  const float* W_k    = (const float*)d_in[6];
  const float* b_k    = (const float*)d_in[7];
  const float* W_v    = (const float*)d_in[8];
  const float* b_v    = (const float*)d_in[9];
  const float* W_q    = (const float*)d_in[10];
  const float* b_q    = (const float*)d_in[11];
  const float* W_i2h  = (const float*)d_in[12];
  const float* W_h2h  = (const float*)d_in[13];
  const float* b_lstm = (const float*)d_in[14];
  const float* Wq_c   = (const float*)d_in[15];
  const float* Wk_c   = (const float*)d_in[16];
  const float* Wv_c   = (const float*)d_in[17];
  const float* Wo_c   = (const float*)d_in[18];
  const float* pi_w   = (const float*)d_in[19];
  const float* pi_b   = (const float*)d_in[20];

  float* probs = (float*)d_out;                         // [B,8,18]
  float* hy    = probs + (size_t)B_ * 144;              // [B,8,256]
  float* cy    = hy + (size_t)B_ * U_ * HID;            // [B,8,256]

  // workspace carve-up (floats)
  float* W = (float*)d_ws;
  size_t off = 0;
  auto alloc = [&](size_t n) { float* p = W + off; off += n; return p; };
  float* h_in  = alloc((size_t)B_ * HID);          // relu(x@W_pre+b)
  float* k0    = alloc((size_t)B_ * 64);           // k_[:,0,:]
  float* v0    = alloc((size_t)B_ * INV);          // v_[:,0,:]
  float* qsel  = alloc((size_t)B_ * U_ * 64);      // option-selected q
  float* sc    = alloc((size_t)B_ * U_);           // scores
  float* coef  = alloc((size_t)B_ * U_);           // sigmoid(s)*mask
  float* maskf = alloc((size_t)B_ * U_);
  float* pre   = alloc((size_t)B_ * U_ * 1024);    // LSTM pre-activations
  float* hb    = alloc((size_t)B_ * U_ * HID);     // h_b (= h_new)
  float* qc    = alloc((size_t)B_ * U_ * 128);
  float* kc    = alloc((size_t)B_ * U_ * 128);
  float* vc    = alloc((size_t)B_ * U_ * 512);
  float* ctx   = alloc((size_t)B_ * U_ * 512);
  float* hsum  = alloc((size_t)B_ * HID);
  float* logit = alloc((size_t)B_ * 144);
  float* Wqr   = alloc((size_t)8 * 256 * 512);
  float* piR   = alloc((size_t)256 * 144);

  // weight reshapes
  prep_wq_kernel<<<4096, 256, 0, stream>>>(W_q, Wqr);
  prep_pi_kernel<<<144, 256, 0, stream>>>(pi_w, piR);

  GemmP p{};
  // K1: h_in = relu(x @ W_pre + b_pre)   [8192,1024]x[1024,256]
  p = GemmP{}; p.A = x; p.lda = OBS; p.Bm = W_pre; p.ldb = HID; p.Out = h_in; p.ldo = HID;
  p.bias = b_pre; p.M = B_; p.N = HID; p.K = OBS;
  gemm_go(EPI_STORE_BIAS_RELU, p, 2, 64, 1, stream);

  // K2a: k0 = h_in @ W_k + b_k   [8192,256]x[256,64]
  p = GemmP{}; p.A = h_in; p.lda = HID; p.Bm = W_k; p.ldb = 64; p.Out = k0; p.ldo = 64;
  p.bias = b_k; p.M = B_; p.N = 64; p.K = HID;
  gemm_go(EPI_STORE_BIAS, p, 1, 64, 1, stream);

  // K2b: v0 = h_in @ W_v + b_v   [8192,256]x[256,400]
  p = GemmP{}; p.A = h_in; p.lda = HID; p.Bm = W_v; p.ldb = INV; p.Out = v0; p.ldo = INV;
  p.bias = b_v; p.M = B_; p.N = INV; p.K = HID;
  gemm_go(EPI_STORE_BIAS, p, 4, 64, 1, stream);

  // K3 (per u): qsel = select_o( hx[:,u,:] @ Wqr[u] + b_q )   N=512, keep 64
  p = GemmP{}; p.A = hx; p.lda = U_ * HID; p.a_zs = HID;
  p.Bm = Wqr; p.ldb = 512; p.b_zs = 256 * 512;
  p.Out = qsel; p.ldo = 512; p.o_zs = 64;
  p.bias = b_q; p.bias_zs = 64; p.opt = option;
  p.M = B_; p.N = 512; p.K = HID;
  gemm_go(EPI_QSEL, p, 4, 64, 8, stream);

  score_kernel<<<(B_ * U_ + 255) / 256, 256, 0, stream>>>(qsel, k0, sc);
  topk_kernel<<<(B_ + 255) / 256, 256, 0, stream>>>(sc, coef, maskf);

  // K5a (per u): pre = coef[b,u] * (v0 @ W_i2h[u])   [8192,400]x[400,1024]
  p = GemmP{}; p.A = v0; p.lda = INV; p.a_zs = 0;
  p.Bm = W_i2h; p.ldb = 1024; p.b_zs = (long)INV * 1024;
  p.Out = pre; p.ldo = U_ * 1024; p.o_zs = 1024;
  p.aux0 = coef; p.M = B_; p.N = 1024; p.K = INV;
  gemm_go(EPI_SCALE, p, 8, 64, 8, stream);

  // K5b (per u): pre += hx[:,u,:] @ W_h2h[u] + b_lstm[u]
  p = GemmP{}; p.A = hx; p.lda = U_ * HID; p.a_zs = HID;
  p.Bm = W_h2h; p.ldb = 1024; p.b_zs = (long)HID * 1024;
  p.Out = pre; p.ldo = U_ * 1024; p.o_zs = 1024;
  p.bias = b_lstm; p.bias_zs = 1024; p.M = B_; p.N = 1024; p.K = HID;
  gemm_go(EPI_ACC_BIAS, p, 8, 64, 8, stream);

  lstm_kernel<<<(int)(((long)B_ * U_ * HID + 255) / 256), 256, 0, stream>>>(pre, cx, maskf, hb, cy);

  // K7 (per u): qc/kc/vc projections of h_b
  p = GemmP{}; p.A = hb; p.lda = U_ * HID; p.a_zs = HID;
  p.Bm = Wq_c; p.ldb = 128; p.b_zs = (long)HID * 128;
  p.Out = qc; p.ldo = U_ * 128; p.o_zs = 128; p.M = B_; p.N = 128; p.K = HID;
  gemm_go(EPI_STORE, p, 1, 64, 8, stream);

  p.Bm = Wk_c; p.Out = kc;
  gemm_go(EPI_STORE, p, 1, 64, 8, stream);

  p = GemmP{}; p.A = hb; p.lda = U_ * HID; p.a_zs = HID;
  p.Bm = Wv_c; p.ldb = 512; p.b_zs = (long)HID * 512;
  p.Out = vc; p.ldo = U_ * 512; p.o_zs = 512; p.M = B_; p.N = 512; p.K = HID;
  gemm_go(EPI_STORE, p, 4, 64, 8, stream);

  attn_kernel<<<B_ / 8, 256, 0, stream>>>(qc, kc, vc, maskf, ctx);

  // K9 (per u): hy = m*(ctx@Wo_c[u] + h_b) + (1-m)*hx  -> straight into d_out
  p = GemmP{}; p.A = ctx; p.lda = U_ * 512; p.a_zs = 512;
  p.Bm = Wo_c; p.ldb = HID; p.b_zs = (long)512 * HID;
  p.Out = hy; p.ldo = U_ * HID; p.o_zs = HID;
  p.aux0 = hb; p.aux1 = hx; p.msk = maskf;
  p.M = B_; p.N = HID; p.K = 512;
  gemm_go(EPI_HCOMM, p, 2, 64, 8, stream);

  hsum_kernel<<<(int)(((long)B_ * HID + 255) / 256), 256, 0, stream>>>(hy, hsum);

  // K10: logits = hsum @ piR + pi_b   [8192,256]x[256,144]
  p = GemmP{}; p.A = hsum; p.lda = HID; p.Bm = piR; p.ldb = 144;
  p.Out = logit; p.ldo = 144; p.bias = pi_b; p.M = B_; p.N = 144; p.K = HID;
  gemm_go(EPI_STORE_BIAS, p, 2, 64, 1, stream);

  softmax18_kernel<<<(B_ * OPT + 255) / 256, 256, 0, stream>>>(logit, probs);
}
